// LSTMPredictor_76759655514632
// MI455X (gfx1250) — matrix operand order repaired and checked by reading, not compiled, not run on hardware
//
#include <hip/hip_runtime.h>

// ---------------------------------------------------------------------------
// Persistent wave32 LSTM on gfx1250 (MI455X).
//   grid: 128 blocks x 128 threads (4 waves); each wave owns 16 batch rows.
//   per step: 2 GEMMs (16x64)@(64x208) via v_wmma_f32_16x16x32_f16 (52 WMMA),
//   biases + scalar-x folded into K rows 51/52 of the f16 weight matrices,
//   gate columns permuted to interleaved order (n' = 4k+gate) so the four
//   gate pre-activations of an element are one float4 LDS load,
//   x streamed 32 steps ahead via global_load_async_to_lds_b128 (ASYNCcnt),
//   B-fragment loads software-pipelined one N-tile ahead,
//   gate nonlinearities via native v_tanh_f32.
// ---------------------------------------------------------------------------

#define HDIM 51
#define GDIM 204          // 4*H
#define NT   13           // 208/16 N tiles
#define TT   2048
#define TOUT 2112         // T + FUTURE

#define HROW    72        // hbuf row stride in f16 (144B -> conflict-free A loads)
#define GSTRIDE 212       // gbuf row stride in f32 (16B-aligned rows, no store conflicts)
#define LSLOT   24        // f16 per B-fragment lane slot (48B, 16B-aligned)

typedef _Float16 v16h __attribute__((ext_vector_type(16)));
typedef _Float16 v8h  __attribute__((ext_vector_type(8)));
typedef float    v8f  __attribute__((ext_vector_type(8)));

// ws dword layout:
//  [0,19968)       52 fragments x 32 lanes x 12 dwords (8 payload + 4 pad)
//  [19968,20019)   W_out (f32), [20019] b_out (f32)
#define FRAG_DWORDS 19968
#define WOUT_OFF    19968

// LDS layout (dynamic, assumed to start at LDS offset 0 - no static __shared__):
//  [0,79872)   weight fragments (48B-strided lane slots, both matrices)
//  [79872,..)  wout[51] + bout
//  per-wave slab (23296B):
//    hbuf f16[16][72] @0      (2304B)
//    gbuf f32[16][212] @2304  (13568B)  gate-interleaved columns: [4k+gate]
//    h2b  f32[16][52] @15872  (3328B)
//    xbuf f32[2][16][32] @19200 (4096B, double-buffered async x stage)
#define SM_WOUT    79872
#define SM_WAVE0   80128
#define HB_OFF     0
#define GB_OFF     2304
#define H2B_OFF    15872
#define XB_OFF     19200
#define WAVE_BYTES 23296
#define SMEM_TOTAL (SM_WAVE0 + 4 * WAVE_BYTES)   // 173312 B < 320KB WGP LDS

__global__ void lstm_setup_kernel(const float* __restrict__ Wih1,
                                  const float* __restrict__ Whh1,
                                  const float* __restrict__ bih1,
                                  const float* __restrict__ bhh1,
                                  const float* __restrict__ Wih2,
                                  const float* __restrict__ Whh2,
                                  const float* __restrict__ bih2,
                                  const float* __restrict__ bhh2,
                                  const float* __restrict__ Wout,
                                  const float* __restrict__ bout,
                                  unsigned* __restrict__ ws) {
  int idx = blockIdx.x * blockDim.x + threadIdx.x;
  if (idx < FRAG_DWORDS) {
    int slot = idx / 12;        // frag*32 + lane
    int r    = idx % 12;
    unsigned uval = 0u;
    if (r < 8) {
      int frag = slot >> 5;     // 0..51
      int lane = slot & 31;
      int mat  = frag / 26;
      int rem  = frag % 26;
      int tile = rem >> 1;
      int kc   = rem & 1;
      int v    = r;
      int ncol = tile * 16 + (lane & 15);   // permuted gate column n' = 4k+gate
      union { _Float16 h[2]; unsigned u; } pk;
      for (int hh = 0; hh < 2; ++hh) {
        // B-fragment mapping (matches A): lanes 0-15 hold K=kc*32+0..7,16..23,
        // lanes 16-31 hold K=kc*32+8..15,24..31, two f16 per dword.
        int k = kc * 32 + ((lane >= 16) ? 8 : 0) + 2 * v + hh + ((v >= 4) ? 8 : 0);
        float val = 0.f;
        if (ncol < GDIM) {
          int gate = ncol & 3;              // 0=i 1=f 2=g 3=o
          int kk   = ncol >> 2;             // hidden index 0..50
          int n    = gate * HDIM + kk;      // original weight row
          if (mat == 0) {
            if (k < HDIM)           val = Whh1[n * HDIM + k];
            else if (k == HDIM)     val = bih1[n] + bhh1[n];   // bias row (A col 51 == 1)
            else if (k == HDIM + 1) val = Wih1[n];             // scalar-x row (A col 52 == x)
          } else {
            if (k < HDIM)           val = Wih2[n * HDIM + k] + Whh2[n * HDIM + k];
            else if (k == HDIM)     val = bih2[n] + bhh2[n];
          }
        }
        pk.h[hh] = (_Float16)val;
      }
      uval = pk.u;
    }
    ws[idx] = uval;
  } else if (idx < WOUT_OFF + HDIM + 1) {
    int k = idx - WOUT_OFF;
    ((float*)ws)[idx] = (k < HDIM) ? Wout[k] : bout[0];
  }
}

__device__ __forceinline__ float tanh_fast(float x) {
#if __has_builtin(__builtin_amdgcn_tanhf)
  return __builtin_amdgcn_tanhf(x);
#else
  float y;
  asm volatile("v_tanh_f32 %0, %1\n\tv_nop" : "=v"(y) : "v"(x));
  return y;
#endif
}
__device__ __forceinline__ float sigmoid_fast(float x) {
  return 0.5f * tanh_fast(0.5f * x) + 0.5f;
}
__device__ __forceinline__ void wait_ds() {
  asm volatile("s_wait_dscnt 0x0" ::: "memory");
}
__device__ __forceinline__ void wait_async() {
  asm volatile("s_wait_asynccnt 0x0" ::: "memory");
}

// Async-stage one 32-step x chunk: x[row0..row0+15][t0..t0+31] -> LDS f32[16][32].
__device__ __forceinline__ void prefetch_x_chunk(const float* __restrict__ xg,
                                                 int row0, int t0,
                                                 unsigned lds_byte_base, int lane) {
#pragma unroll
  for (int q = 0; q < 4; ++q) {
    int seg = lane + 32 * q;
    int r   = seg >> 3;         // batch row within tile
    int si  = seg & 7;          // 4-float segment within the 32-step row
    unsigned voff = (unsigned)((((row0 + r) * TT) + t0 + si * 4) * 4);   // global byte offset
    unsigned dst  = lds_byte_base + (unsigned)((r * 32 + si * 4) * 4);   // LDS byte address
    asm volatile("global_load_async_to_lds_b128 %0, %1, %2 offset:0"
                 :: "v"(dst), "v"(voff), "s"(xg) : "memory");
  }
}

// A fragment (16x32 f16, MxK) from row-major 16xHROW f16 LDS tile (ISA 7.12.2)
__device__ __forceinline__ v16h load_afrag(const _Float16* hb, int lane, int kc) {
  int off = (lane & 15) * HROW + kc * 32 + ((lane >> 4) << 3);
  v8h lo = *(const v8h*)(hb + off);        // K 0..7   (upper half: 8..15)
  v8h hi = *(const v8h*)(hb + off + 16);   // K 16..23 (upper half: 24..31)
  v16h a;
#pragma unroll
  for (int i = 0; i < 8; ++i) { a[i] = lo[i]; a[i + 8] = hi[i]; }
  return a;
}

// B fragment, pre-swizzled at setup: 16 f16 payload in a 24-f16 (48B) lane slot
__device__ __forceinline__ v16h load_bfrag(const _Float16* wl, int frag, int lane) {
  const _Float16* p = wl + (frag * 32 + lane) * LSLOT;
  v8h lo = *(const v8h*)p;
  v8h hi = *(const v8h*)(p + 8);
  v16h b;
#pragma unroll
  for (int i = 0; i < 8; ++i) { b[i] = lo[i]; b[i + 8] = hi[i]; }
  return b;
}

// gates[16][208] = [h | 1 | x | 0pad] @ W'; B-fragment loads pipelined 1 tile ahead
__device__ __forceinline__ void layer_matmul(const _Float16* wl, int matbase,
                                             const _Float16* hb, float* gb, int lane) {
  v16h a0 = load_afrag(hb, lane, 0);
  v16h a1 = load_afrag(hb, lane, 1);
  int crow = (lane >> 4) * 8;   // C/D layout: vgpr r holds M=r(+8), lane%16 = N
  int ccol = lane & 15;
  v16h b0 = load_bfrag(wl, matbase + 0, lane);
  v16h b1 = load_bfrag(wl, matbase + 1, lane);
#pragma unroll
  for (int t = 0; t < NT; ++t) {
    int tn = (t + 1 < NT) ? (t + 1) : t;           // last iter: harmless reload
    v16h nb0 = load_bfrag(wl, matbase + tn * 2 + 0, lane);
    v16h nb1 = load_bfrag(wl, matbase + tn * 2 + 1, lane);
    v8f acc = {};
    acc = __builtin_amdgcn_wmma_f32_16x16x32_f16(false, a0, false, b0, (short)0, acc, false, false);
    acc = __builtin_amdgcn_wmma_f32_16x16x32_f16(false, a1, false, b1, (short)0, acc, false, false);
    float* g = gb + t * 16 + ccol;
#pragma unroll
    for (int r = 0; r < 8; ++r) g[(crow + r) * GSTRIDE] = acc[r];
    b0 = nb0; b1 = nb1;
  }
}

__launch_bounds__(128)
__global__ void lstm_kernel(const float* __restrict__ xg,
                            float* __restrict__ outg,
                            const unsigned* __restrict__ ws) {
  extern __shared__ char smem[];
  const int tid  = threadIdx.x;
  const int wid  = tid >> 5;
  const int lane = tid & 31;

  // cooperative load: weights + wout into LDS
  {
    const uint4* src = (const uint4*)ws;
    uint4* dst = (uint4*)smem;
    for (int i = tid; i < FRAG_DWORDS / 4; i += 128) dst[i] = src[i];
    float* wof = (float*)(smem + SM_WOUT);
    for (int i = tid; i < HDIM + 1; i += 128) wof[i] = ((const float*)ws)[WOUT_OFF + i];
  }

  const _Float16* wl = (const _Float16*)smem;
  const float*    wo = (const float*)(smem + SM_WOUT);
  const unsigned wavebase = (unsigned)(SM_WAVE0 + wid * WAVE_BYTES);
  _Float16* hb  = (_Float16*)(smem + wavebase + HB_OFF);
  float*    gb  = (float*)(smem + wavebase + GB_OFF);
  float*    h2b = (float*)(smem + wavebase + H2B_OFF);
  float*    xb  = (float*)(smem + wavebase + XB_OFF);
  const unsigned xlds = wavebase + XB_OFF;   // LDS byte address (dynamic LDS at offset 0)

  const int row0 = blockIdx.x * 64 + wid * 16;

  // stage first two x chunks (t=0..31 -> buf0, t=32..63 -> buf1)
  prefetch_x_chunk(xg, row0, 0, xlds, lane);
  prefetch_x_chunk(xg, row0, 32, xlds + 2048, lane);

  // h = 0; bias column (51) = 1
  for (int i = lane; i < 16 * HROW; i += 32) hb[i] = (_Float16)0.f;
  wait_async();
  wait_ds();
  if (lane < 16) {
    hb[lane * HROW + 51] = (_Float16)1.0f;
    hb[lane * HROW + 52] = (_Float16)xb[lane * 32];   // x[:,0]
  }
  __syncthreads();

  float c[26];
#pragma unroll
  for (int j = 0; j < 26; ++j) c[j] = 0.f;
  const float boutv = wo[HDIM];
  int curbuf = 0;

  for (int step = 0; step < TOUT; ++step) {
    // chunk boundary: next consumed x crosses into the other buffer
    if (((step & 31) == 31) && (step + 1 < TT)) {
      wait_async();
      curbuf ^= 1;
      int tnext = step + 1 + 32;
      if (tnext < TT)
        prefetch_x_chunk(xg, row0, tnext, xlds + (unsigned)((curbuf ^ 1) * 2048), lane);
    }
    const float* xcur = xb + curbuf * 512;

    // ---- layer 1 GEMM ----
    layer_matmul(wl, 0, hb, gb, lane);
    wait_ds();

    // ---- layer 1 elementwise: c,h update (816 elems, 26 per lane) ----
    {
      int e = lane, m = 0, k = lane;
#pragma unroll
      for (int j = 0; j < 26; ++j) {
        if (e < 816) {
          const float4 gv = *(const float4*)(gb + m * GSTRIDE + 4 * k);  // i,f,g,o
          float i_ = sigmoid_fast(gv.x);
          float f_ = sigmoid_fast(gv.y);
          float g_ = tanh_fast(gv.z);
          float o_ = sigmoid_fast(gv.w);
          float cn = f_ * c[j] + i_ * g_;
          c[j] = cn;
          hb[m * HROW + k] = (_Float16)(o_ * tanh_fast(cn));
        }
        e += 32; k += 32;
        if (k >= 51) { k -= 51; m += 1; }
      }
    }
    // x for next step from the LDS stage (layer-1 A already consumed col 52;
    // layer-2's weight row 52 is zero, so this store is safe here)
    if ((step + 1 < TT) && (lane < 16))
      hb[lane * HROW + 52] = (_Float16)xcur[lane * 32 + ((step + 1) & 31)];
    wait_ds();

    // ---- layer 2 GEMM (stateless cell, combined W) ----
    layer_matmul(wl, 26, hb, gb, lane);
    wait_ds();

    // ---- layer 2 elementwise -> h2 ----
    {
      int e = lane, m = 0, k = lane;
#pragma unroll
      for (int j = 0; j < 26; ++j) {
        if (e < 816) {
          const float4 gv = *(const float4*)(gb + m * GSTRIDE + 4 * k);  // i,f,g,o
          float i_ = sigmoid_fast(gv.x);
          float f_ = sigmoid_fast(gv.y);
          float g_ = tanh_fast(gv.z);
          float o_ = sigmoid_fast(gv.w);
          float c2 = f_ * c[j] + i_ * g_;
          h2b[m * 52 + k] = o_ * tanh_fast(c2);
        }
        e += 32; k += 32;
        if (k >= 51) { k -= 51; m += 1; }
      }
    }
    wait_ds();

    // ---- emit: out = h2 . w_out + b_out (split over 32 lanes + shfl) ----
    {
      int r  = lane & 15;
      int hh = lane >> 4;
      int k0 = hh * 26;
      const float* hr = h2b + r * 52 + k0;
      float s = 0.f;
#pragma unroll
      for (int i = 0; i < 26; ++i)
        if (k0 + i < HDIM) s += hr[i] * wo[k0 + i];
      s += __shfl_xor(s, 16, 32);
      if (lane < 16) {
        s += boutv;
        outg[(size_t)(row0 + r) * TOUT + step] = s;
        if (step + 1 >= TT) hb[r * HROW + 52] = (_Float16)s;   // autoregressive input
      }
    }
    wait_ds();
  }
}

extern "C" void kernel_launch(void* const* d_in, const int* in_sizes, int n_in,
                              void* d_out, int out_size, void* d_ws, size_t ws_size,
                              hipStream_t stream) {
  (void)in_sizes; (void)n_in; (void)out_size; (void)ws_size;
  const float* x    = (const float*)d_in[0];
  const float* Wih1 = (const float*)d_in[1];
  const float* Whh1 = (const float*)d_in[2];
  const float* bih1 = (const float*)d_in[3];
  const float* bhh1 = (const float*)d_in[4];
  const float* Wih2 = (const float*)d_in[5];
  const float* Whh2 = (const float*)d_in[6];
  const float* bih2 = (const float*)d_in[7];
  const float* bhh2 = (const float*)d_in[8];
  const float* Wout = (const float*)d_in[9];
  const float* bout = (const float*)d_in[10];
  float* out = (float*)d_out;
  unsigned* ws = (unsigned*)d_ws;

  lstm_setup_kernel<<<79, 256, 0, stream>>>(Wih1, Whh1, bih1, bhh1,
                                            Wih2, Whh2, bih2, bhh2,
                                            Wout, bout, ws);

  (void)hipFuncSetAttribute((const void*)lstm_kernel,
                            hipFuncAttributeMaxDynamicSharedMemorySize,
                            (int)SMEM_TOTAL);
  lstm_kernel<<<128, 128, SMEM_TOTAL, stream>>>(x, out, ws);
}
